// GenProjection_56831007260833
// MI455X (gfx1250) — matrix operand order repaired and checked by reading, not compiled, not run on hardware
//
#include <hip/hip_runtime.h>
#include <stdint.h>

#define B_ 4
#define T_ 128
#define M_ 512           // B*T
#define H_ 1024
#define V_ 50265
#define N_ 300
#define E_ 4800

typedef float v8f   __attribute__((ext_vector_type(8)));
typedef float f32x4 __attribute__((ext_vector_type(4)));
typedef __bf16 v16bf __attribute__((ext_vector_type(16)));

union BFrag { v16bf v; f32x4 f[2]; };

__device__ __forceinline__ unsigned short f2bf(float x) {
  uint32_t u = __float_as_uint(x);
  u += 0x7FFFu + ((u >> 16) & 1u);          // round-to-nearest-even
  return (unsigned short)(u >> 16);
}

__device__ __forceinline__ float sigmoidf(float x) {
  return 1.0f / (1.0f + __expf(-x));
}

// ---------------------------------------------------------------------------
// Convert fp32 activations into bf16 A-matrices: hq=[hs|qt], qe=[qt|te],
// fa=[hs|qt|te] (for the 3-segment gen_feat GEMM).
// ---------------------------------------------------------------------------
__global__ void convert_kernel(const float* __restrict__ hs, const float* __restrict__ qt,
                               const float* __restrict__ te,
                               unsigned short* __restrict__ hqb, unsigned short* __restrict__ qeb,
                               unsigned short* __restrict__ fa) {
  int row = blockIdx.x;
  const float* h = hs + (size_t)row * H_;
  const float* q = qt + (size_t)row * H_;
  const float* e = te + (size_t)row * H_;
  unsigned short* hq = hqb + (size_t)row * 2048;
  unsigned short* qe = qeb + (size_t)row * 2048;
  unsigned short* f  = fa  + (size_t)row * 3072;
  for (int i = threadIdx.x; i < H_; i += blockDim.x) {
    unsigned short hb = f2bf(h[i]), qb = f2bf(q[i]), eb = f2bf(e[i]);
    hq[i] = hb; hq[1024 + i] = qb;
    qe[i] = qb; qe[1024 + i] = eb;
    f[i] = hb;  f[1024 + i] = qb; f[2048 + i] = eb;
  }
}

// ---------------------------------------------------------------------------
// Generic bf16 WMMA GEMM: out[M=512, Ncols] = A[512,K]bf16 @ W(f32,k-major,ldw)
// Weight selected from up to 3 row-stacked segments of height segK.
// Epilogue: (+bias[col]) and optionally *(1 - psScale[row]).
// Workgroup = 8 waves; covers a 32-wide N stripe and all 512 rows.
// ---------------------------------------------------------------------------
#define LDSK 40   // LDS row stride in bf16 units for a 32-deep K tile

__global__ void __launch_bounds__(256) gemm_bf16_kernel(
    const unsigned short* __restrict__ A, int lda, int K,
    const float* __restrict__ W0, const float* __restrict__ W1, const float* __restrict__ W2,
    int segK, int ldw, int Ncols,
    const float* __restrict__ bias, const float* __restrict__ psScale,
    float* __restrict__ out, int ldo) {
  __shared__ unsigned short Bt[32 * LDSK];
  const int tid  = threadIdx.x;
  const int wave = tid >> 5;
  const int lane = tid & 31;
  const int n0   = blockIdx.x * 32;
  const int h    = lane >> 4;   // half-wave id
  const int nl   = lane & 15;

  v8f acc[4][2];
  for (int mt = 0; mt < 4; ++mt)
    for (int nt = 0; nt < 2; ++nt)
      for (int r = 0; r < 8; ++r) acc[mt][nt][r] = 0.0f;

  // LDS-fill mapping: thread owns column (n0+fn), four consecutive k at fk.
  const int fn = tid & 31;
  const int fk = (tid >> 5) * 4;
  const int mrow0 = wave * 64;

  for (int k0 = 0; k0 < K; k0 += 32) {
    // Uniform weight-segment select
    const float* W = W0; int kr = k0;
    if (k0 >= 2 * segK)      { W = W2; kr = k0 - 2 * segK; }
    else if (k0 >= segK)     { W = W1; kr = k0 - segK; }

    __syncthreads();
    // Cooperative fp32 load + bf16 convert into LDS tile Bt[n][k]
    {
      int gcol = n0 + fn;
      float w0 = 0.f, w1 = 0.f, w2 = 0.f, w3 = 0.f;
      if (gcol < Ncols) {
        const float* wp = W + (size_t)(kr + fk) * (size_t)ldw + gcol;
        w0 = wp[0];
        w1 = wp[(size_t)ldw];
        w2 = wp[2 * (size_t)ldw];
        w3 = wp[3 * (size_t)ldw];
      }
      uint32_t p0 = (uint32_t)f2bf(w0) | ((uint32_t)f2bf(w1) << 16);
      uint32_t p1 = (uint32_t)f2bf(w2) | ((uint32_t)f2bf(w3) << 16);
      uint32_t* dst = (uint32_t*)&Bt[fn * LDSK + fk];
      dst[0] = p0; dst[1] = p1;
    }
    __syncthreads();

    // B fragments: lane holds 16 contiguous K (K = h*16..h*16+15), col = nl.
    BFrag bf0, bf1;
    {
      const unsigned short* r0 = &Bt[nl * LDSK + h * 16];
      bf0.f[0] = *(const f32x4*)(r0);
      bf0.f[1] = *(const f32x4*)(r0 + 8);
      const unsigned short* r1 = &Bt[(16 + nl) * LDSK + h * 16];
      bf1.f[0] = *(const f32x4*)(r1);
      bf1.f[1] = *(const f32x4*)(r1 + 8);
    }

    // A fragments: lane m = nl; 16 contiguous K at h*16 (two b128 loads).
    for (int mt = 0; mt < 4; ++mt) {
      int row = mrow0 + mt * 16 + nl;
      BFrag af;
      const unsigned short* ap = A + (size_t)row * (size_t)lda + k0 + h * 16;
      af.f[0] = *(const f32x4*)(ap);
      af.f[1] = *(const f32x4*)(ap + 8);
      acc[mt][0] = __builtin_amdgcn_wmma_f32_16x16x32_bf16(
          false, af.v, false, bf0.v, (short)0, acc[mt][0], false, false);
      acc[mt][1] = __builtin_amdgcn_wmma_f32_16x16x32_bf16(
          false, af.v, false, bf1.v, (short)0, acc[mt][1], false, false);
    }
  }

  // Epilogue: C VGPR r -> M = r + 8*h; N = nl.
  for (int nt = 0; nt < 2; ++nt) {
    int col = n0 + nt * 16 + nl;
    if (col >= Ncols) continue;
    float bv = bias ? bias[col] : 0.0f;
    for (int mt = 0; mt < 4; ++mt) {
      int rbase = mrow0 + mt * 16 + h * 8;
      for (int r = 0; r < 8; ++r) {
        int row = rbase + r;
        float val = acc[mt][nt][r] + bv;
        if (psScale) val *= (1.0f - psScale[row]);
        out[(size_t)row * (size_t)ldo + col] = val;
      }
    }
  }
}

// ---------------------------------------------------------------------------
// Per-row gates: pro_switch, gsem_gate, memory_gate (block-reduced dots).
// ---------------------------------------------------------------------------
__device__ __forceinline__ float blockReduceSum(float v, float* red) {
  int tid = threadIdx.x;
  red[tid] = v; __syncthreads();
  for (int s = 128; s > 0; s >>= 1) { if (tid < s) red[tid] += red[tid + s]; __syncthreads(); }
  float r = red[0]; __syncthreads();
  return r;
}

__global__ void __launch_bounds__(256) scalars_kernel(
    const float* __restrict__ gf, const float* __restrict__ hs, const float* __restrict__ qt,
    const float* __restrict__ Wps, const float* __restrict__ bps,
    const float* __restrict__ Wgg, const float* __restrict__ bgg,
    const float* __restrict__ Wmg, const float* __restrict__ bmg,
    float* __restrict__ ps, float* __restrict__ gg, float* __restrict__ mg,
    float* __restrict__ ps_out) {
  __shared__ float red[256];
  int row = blockIdx.x, tid = threadIdx.x;
  const float* g = gf + (size_t)row * H_;
  const float* h = hs + (size_t)row * H_;
  const float* q = qt + (size_t)row * H_;
  float s1 = 0.f, s2 = 0.f, s3 = 0.f;
  for (int k = tid; k < H_; k += 256) {
    float hv = h[k], qv = q[k];
    s1 += g[k] * Wps[k];
    s2 += hv * Wgg[k] + qv * Wgg[H_ + k];
    s3 += hv * Wmg[k] + qv * Wmg[H_ + k];
  }
  s1 = blockReduceSum(s1, red);
  s2 = blockReduceSum(s2, red);
  s3 = blockReduceSum(s3, red);
  if (tid == 0) {
    float p = sigmoidf(s1 + bps[0]);
    ps[row] = p; ps_out[row] = p;
    gg[row] = sigmoidf(s2 + bgg[0]);
    mg[row] = sigmoidf(s3 + bmg[0]);
  }
}

// ---------------------------------------------------------------------------
__global__ void zero2_kernel(float* a, int na, float* b, int nb) {
  int i = blockIdx.x * blockDim.x + threadIdx.x;
  if (i < na) a[i] = 0.f;
  if (i < nb) b[i] = 0.f;
}

__global__ void edges_kernel(const int* __restrict__ edges, float* __restrict__ Graph) {
  int i = blockIdx.x * blockDim.x + threadIdx.x;
  if (i >= B_ * E_) return;
  int b = i / E_, e = i % E_;
  int src = edges[(size_t)b * 2 * E_ + e];
  int dst = edges[(size_t)b * 2 * E_ + E_ + e];
  atomicAdd(Graph + (size_t)b * N_ * N_ + (size_t)src * N_ + dst, 1.0f);
}

// ---------------------------------------------------------------------------
// Sequential gated-memory recurrence + graph matvec (one block per batch).
// ---------------------------------------------------------------------------
__global__ void __launch_bounds__(320) recur_kernel(
    const float* __restrict__ cp, const float* __restrict__ mg,
    const float* __restrict__ Graph, float* __restrict__ gsem) {
  __shared__ float m[N_];
  int b = blockIdx.x, i = threadIdx.x;
  if (i < N_) m[i] = 0.f;
  __syncthreads();
  const float* G = Graph + (size_t)b * N_ * N_;
  for (int t = 0; t < T_; ++t) {
    int row = b * T_ + t;
    float mgv = mg[row];
    if (i < N_) m[i] = cp[(size_t)row * N_ + i] * mgv + m[i] * (1.f - mgv);
    __syncthreads();
    if (i < N_) {
      const float* Gr = G + (size_t)i * N_;
      float gv = 0.f;
      for (int j = 0; j < N_; ++j) gv += Gr[j] * m[j];
      gsem[(size_t)row * N_ + i] = gv;
    }
    __syncthreads();
  }
}

// ---------------------------------------------------------------------------
// pro_gsem, softmax->coverage, and node->vocab scatter (one block per row).
// ---------------------------------------------------------------------------
__global__ void __launch_bounds__(256) final_kernel(
    const float* __restrict__ cp, const float* __restrict__ gsem,
    const float* __restrict__ ps, const float* __restrict__ gg,
    const int* __restrict__ map, float* __restrict__ total, float* __restrict__ cov) {
  __shared__ float red[256];
  int row = blockIdx.x, tid = threadIdx.x;
  int b = row / T_;
  float ggv = gg[row], psv = ps[row];
  int n0 = tid, n1 = tid + 256;
  float pg0 = 0.f, pg1 = 0.f, l0 = -1e30f, l1 = -1e30f;
  if (n0 < N_) { pg0 = cp[(size_t)row * N_ + n0] * ggv + gsem[(size_t)row * N_ + n0] * (1.f - ggv); l0 = pg0 * psv; }
  if (n1 < N_) { pg1 = cp[(size_t)row * N_ + n1] * ggv + gsem[(size_t)row * N_ + n1] * (1.f - ggv); l1 = pg1 * psv; }

  red[tid] = fmaxf(l0, l1); __syncthreads();
  for (int s = 128; s > 0; s >>= 1) { if (tid < s) red[tid] = fmaxf(red[tid], red[tid + s]); __syncthreads(); }
  float mx = red[0]; __syncthreads();

  float e0 = (n0 < N_) ? __expf(l0 - mx) : 0.f;
  float e1 = (n1 < N_) ? __expf(l1 - mx) : 0.f;
  red[tid] = e0 + e1; __syncthreads();
  for (int s = 128; s > 0; s >>= 1) { if (tid < s) red[tid] += red[tid + s]; __syncthreads(); }
  float inv = 1.0f / red[0];

  if (n0 < N_) {
    atomicAdd(cov + (size_t)b * N_ + n0, e0 * inv);
    atomicAdd(total + (size_t)row * V_ + map[n0], pg0 * psv);
  }
  if (n1 < N_) {
    atomicAdd(cov + (size_t)b * N_ + n1, e1 * inv);
    atomicAdd(total + (size_t)row * V_ + map[n1], pg1 * psv);
  }
}

// ---------------------------------------------------------------------------
extern "C" void kernel_launch(void* const* d_in, const int* in_sizes, int n_in,
                              void* d_out, int out_size, void* d_ws, size_t ws_size,
                              hipStream_t stream) {
  (void)in_sizes; (void)n_in; (void)out_size; (void)ws_size;

  const float* hs  = (const float*)d_in[0];
  const float* qt  = (const float*)d_in[1];
  const float* te  = (const float*)d_in[2];
  const int*   edg = (const int*)d_in[3];
  const int*   map = (const int*)d_in[4];
  const float* Wg  = (const float*)d_in[5];
  const float* bg  = (const float*)d_in[6];
  const float* Wh  = (const float*)d_in[7];
  const float* Ws  = (const float*)d_in[8];
  const float* Wx  = (const float*)d_in[9];
  const float* bx  = (const float*)d_in[10];
  const float* Wps = (const float*)d_in[11];
  const float* bps = (const float*)d_in[12];
  const float* Wc  = (const float*)d_in[13];
  const float* bc  = (const float*)d_in[14];
  const float* Wgg = (const float*)d_in[15];
  const float* bgg = (const float*)d_in[16];
  const float* Wmg = (const float*)d_in[17];
  const float* bmg = (const float*)d_in[18];

  // workspace layout (~11.6 MB total)
  char* ws = (char*)d_ws;
  unsigned short* hqb = (unsigned short*)(ws + 0);          // 512x2048 bf16
  unsigned short* qeb = (unsigned short*)(ws + 2097152);    // 512x2048 bf16
  unsigned short* fa  = (unsigned short*)(ws + 4194304);    // 512x3072 bf16
  float* gen_feat = (float*)(ws + 7340032);                 // 512x1024 f32
  float* ps   = (float*)(ws + 9437184);                     // 512
  float* gg   = (float*)(ws + 9439232);                     // 512
  float* mg   = (float*)(ws + 9441280);                     // 512
  float* Grph = (float*)(ws + 9443328);                     // 4x300x300
  float* cp   = (float*)(ws + 10883328);                    // 512x300
  float* gsm  = (float*)(ws + 11497728);                    // 512x300

  float* total  = (float*)d_out;                 // [B,T,V]
  float* cov    = total + (size_t)M_ * V_;       // [B,N]
  float* ps_out = cov + B_ * N_;                 // [B,T]

  convert_kernel<<<M_, 256, 0, stream>>>(hs, qt, te, hqb, qeb, fa);

  int nz = B_ * N_ * N_;
  zero2_kernel<<<(nz + 255) / 256, 256, 0, stream>>>(Grph, nz, cov, B_ * N_);
  edges_kernel<<<(B_ * E_ + 255) / 256, 256, 0, stream>>>(edg, Grph);

  // gen_feat = [hs|qt|te] @ [Wh;Ws;Wx] + bx
  gemm_bf16_kernel<<<H_ / 32, 256, 0, stream>>>(fa, 3 * H_, 3 * H_, Wh, Ws, Wx,
                                                H_, H_, H_, bx, nullptr, gen_feat, H_);

  scalars_kernel<<<M_, 256, 0, stream>>>(gen_feat, hs, qt, Wps, bps, Wgg, bgg,
                                         Wmg, bmg, ps, gg, mg, ps_out);

  // copy_pro = [qt|te] @ Wc + bc
  gemm_bf16_kernel<<<(N_ + 31) / 32, 256, 0, stream>>>(qeb, 2 * H_, 2 * H_, Wc, nullptr, nullptr,
                                                       2 * H_, N_, N_, bc, nullptr, cp, N_);

  // total = ([hs|qt] @ Wg + bg) * (1 - ps)   (dominant GEMM, fused epilogue)
  gemm_bf16_kernel<<<(V_ + 31) / 32, 256, 0, stream>>>(hqb, 2 * H_, 2 * H_, Wg, nullptr, nullptr,
                                                       2 * H_, V_, V_, bg, ps, total, V_);

  recur_kernel<<<B_, 320, 0, stream>>>(cp, mg, Grph, gsm);

  final_kernel<<<M_, 256, 0, stream>>>(cp, gsm, ps, gg, map, total, cov);
}
